// InternalMemory_74406013436033
// MI455X (gfx1250) — matrix operand order repaired and checked by reading, not compiled, not run on hardware
//
#include <hip/hip_runtime.h>
#include <math.h>

// Problem constants (from reference setup_inputs)
#define DD   1024
#define NN   8192          // B*L = 4*2048 tokens
#define SS   4096
#define KTOP 8
#define EPSF 1e-8f

typedef __attribute__((ext_vector_type(16))) __bf16 v16bf;
typedef __attribute__((ext_vector_type(8)))  float  v8f;

union BF16Frag {
  v16bf v;
  unsigned short u[16];
  unsigned int   w[8];
  uint4          q[2];
};

__device__ __forceinline__ unsigned short f2bf(float f) {
  union { float f; unsigned int u; } a; a.f = f;
  unsigned int r = a.u + 0x7FFFu + ((a.u >> 16) & 1u);   // round-nearest-even
  return (unsigned short)(r >> 16);
}
__device__ __forceinline__ float bf2f(unsigned short h) {
  union { unsigned int u; float f; } a; a.u = ((unsigned int)h) << 16;
  return a.f;
}

// ---------------------------------------------------------------------------
// 1) Convert Wr/Wi to bf16 (and a pre-negated Wi so the GEMM needs no VALU xor)
__global__ void conv_w_kernel(const float* __restrict__ Wr, const float* __restrict__ Wi,
                              unsigned short* __restrict__ wr_bf, unsigned short* __restrict__ wi_bf,
                              unsigned short* __restrict__ win_bf, int n) {
  int i = blockIdx.x * blockDim.x + threadIdx.x;
  if (i < n) {
    wr_bf[i] = f2bf(Wr[i]);
    unsigned short b = f2bf(Wi[i]);
    wi_bf[i]  = b;
    win_bf[i] = b ^ 0x8000u;
  }
}

// 2) Split keys [S,D,2] into bf16 k_r/k_i and compute k_mag[s]
__global__ __launch_bounds__(256) void conv_keys_kernel(const float* __restrict__ keys,
                                 unsigned short* __restrict__ kr_bf, unsigned short* __restrict__ ki_bf,
                                 float* __restrict__ kmag) {
  const int s = blockIdx.x, tid = threadIdx.x;
  const float* row = keys + (size_t)s * (DD * 2);
  float acc = 0.f;
#pragma unroll
  for (int j = 0; j < DD / 256; ++j) {
    int d = tid + 256 * j;
    float2 v = *(const float2*)(row + 2 * d);
    kr_bf[(size_t)s * DD + d] = f2bf(v.x);
    ki_bf[(size_t)s * DD + d] = f2bf(v.y);
    acc += v.x * v.x + v.y * v.y;
  }
  __shared__ float red[256];
  red[tid] = acc; __syncthreads();
  for (int off = 128; off > 0; off >>= 1) {
    if (tid < off) red[tid] += red[tid + off];
    __syncthreads();
  }
  if (tid == 0) kmag[s] = sqrtf(red[0] + EPSF);
}

// 3) De-interleave x [N,D,2] into bf16 x_r/x_i (read once, used 16x by GEMM)
__global__ __launch_bounds__(256) void conv_x_kernel(const float* __restrict__ x,
                              unsigned short* __restrict__ xr_bf, unsigned short* __restrict__ xi_bf) {
  const size_t i = (size_t)blockIdx.x * 256 + threadIdx.x;   // over N*D
  float2 v = *(const float2*)(x + 2 * i);
  xr_bf[i] = f2bf(v.x);
  xi_bf[i] = f2bf(v.y);
}

// ---------------------------------------------------------------------------
// 4) Complex linear via WMMA bf16:
//    q_r = x_r*Wr^T + x_i*(-Wi)^T ; q_i = x_r*Wi^T + x_i*Wr^T
//    Each wave: 16 tokens x 64 outputs. Writes q in bf16.
//    WMMAs ordered to alternate accumulators (accR/accI) so dependent
//    matrix ops are never back-to-back (avoids the 4-coexec hazard NOPs).
__global__ __launch_bounds__(256) void gemm_q_kernel(
    const unsigned short* __restrict__ xr_bf, const unsigned short* __restrict__ xi_bf,
    const unsigned short* __restrict__ wr_bf, const unsigned short* __restrict__ wi_bf,
    const unsigned short* __restrict__ win_bf,
    unsigned short* __restrict__ qr_bf, unsigned short* __restrict__ qi_bf) {
  const int lane   = threadIdx.x & 31;
  const int wave   = threadIdx.x >> 5;
  const int w      = blockIdx.x * 8 + wave;       // 8192 waves total
  const int mtile  = w >> 4;                      // 512 token tiles
  const int egrp   = w & 15;                      // 16 groups of 64 outputs
  const int tokBase = mtile * 16;
  const int eBase   = egrp * 64;
  const int lrow = lane & 15;
  const bool hi  = lane >= 16;
  const int akoff = hi ? 16 : 0;                  // A-frag K base (16x32 bf16 layout)
  const int bkoff = hi ? 16 : 0;                  // B-frag K base (32x16 bf16 layout)

  v8f accR[4] = {}, accI[4] = {};
  const size_t arow = (size_t)(tokBase + lrow) * DD;

  for (int k0 = 0; k0 < DD; k0 += 32) {
    BF16Frag ar, ai;
    const int kb = k0 + akoff;
    ar.q[0] = *(const uint4*)(xr_bf + arow + kb);
    ar.q[1] = *(const uint4*)(xr_bf + arow + kb + 8);
    ai.q[0] = *(const uint4*)(xi_bf + arow + kb);
    ai.q[1] = *(const uint4*)(xi_bf + arow + kb + 8);
#pragma unroll
    for (int t = 0; t < 4; ++t) {
      const int col = eBase + t * 16 + lrow;
      const size_t boff = (size_t)col * DD + (size_t)(k0 + bkoff);
      BF16Frag bwr, bwi, bwn;
      bwr.q[0] = ((const uint4*)(wr_bf + boff))[0];
      bwr.q[1] = ((const uint4*)(wr_bf + boff))[1];
      bwi.q[0] = ((const uint4*)(wi_bf + boff))[0];
      bwi.q[1] = ((const uint4*)(wi_bf + boff))[1];
      bwn.q[0] = ((const uint4*)(win_bf + boff))[0];
      bwn.q[1] = ((const uint4*)(win_bf + boff))[1];
      // speculative prefetch (SE scope) -- invalid tail addresses are dropped
      __builtin_prefetch(wr_bf + boff + 32, 0, 3);
      __builtin_prefetch(wi_bf + boff + 32, 0, 3);
      __builtin_prefetch(win_bf + boff + 32, 0, 3);
      // alternate accumulators: no two consecutive WMMAs share a destination
      accR[t] = __builtin_amdgcn_wmma_f32_16x16x32_bf16(false, ar.v, false, bwr.v, (short)0, accR[t], false, false);
      accI[t] = __builtin_amdgcn_wmma_f32_16x16x32_bf16(false, ar.v, false, bwi.v, (short)0, accI[t], false, false);
      accR[t] = __builtin_amdgcn_wmma_f32_16x16x32_bf16(false, ai.v, false, bwn.v, (short)0, accR[t], false, false);
      accI[t] = __builtin_amdgcn_wmma_f32_16x16x32_bf16(false, ai.v, false, bwr.v, (short)0, accI[t], false, false);
    }
  }
  // C layout: VGPR r of lane l -> M = r + 8*(l>=16), N = l&15
  const int moff = hi ? 8 : 0;
#pragma unroll
  for (int t = 0; t < 4; ++t) {
    const int col = eBase + t * 16 + lrow;
#pragma unroll
    for (int r = 0; r < 8; ++r) {
      const size_t o = (size_t)(tokBase + moff + r) * DD + col;
      qr_bf[o] = f2bf(accR[t][r]);
      qi_bf[o] = f2bf(accI[t][r]);
    }
  }
}

// 5) q_mag[n] = sqrt(sum |q|^2 + eps)
__global__ __launch_bounds__(256) void qmag_kernel(const unsigned short* __restrict__ qr_bf,
                            const unsigned short* __restrict__ qi_bf, float* __restrict__ qmag) {
  const int n = blockIdx.x, tid = threadIdx.x;
  float acc = 0.f;
#pragma unroll
  for (int j = 0; j < DD / 256; ++j) {
    int d = tid + 256 * j;
    float r = bf2f(qr_bf[(size_t)n * DD + d]);
    float i = bf2f(qi_bf[(size_t)n * DD + d]);
    acc += r * r + i * i;
  }
  __shared__ float red[256];
  red[tid] = acc; __syncthreads();
  for (int off = 128; off > 0; off >>= 1) {
    if (tid < off) red[tid] += red[tid + off];
    __syncthreads();
  }
  if (tid == 0) qmag[n] = sqrtf(red[0] + EPSF);
}

// ---------------------------------------------------------------------------
// 6) Score GEMM via WMMA: dot = q_r k_r^T + q_i k_i^T, scaled by 1/(|q||k|+eps)
//    t-tiles processed pairwise so dependent accumulates are >=2 apart.
__global__ __launch_bounds__(256) void gemm_scores_kernel(
    const unsigned short* __restrict__ qr_bf, const unsigned short* __restrict__ qi_bf,
    const unsigned short* __restrict__ kr_bf, const unsigned short* __restrict__ ki_bf,
    const float* __restrict__ qmag, const float* __restrict__ kmag,
    float* __restrict__ scores) {
  const int lane  = threadIdx.x & 31;
  const int wave  = threadIdx.x >> 5;
  const int w     = blockIdx.x * 8 + wave;        // 32768 waves total
  const int mtile = w >> 6;                       // 512 token tiles
  const int sgrp  = w & 63;                       // 64 groups of 64 keys
  const int tokBase = mtile * 16;
  const int sBase   = sgrp * 64;
  const int lrow = lane & 15;
  const bool hi  = lane >= 16;
  const int akoff = hi ? 16 : 0;
  const int bkoff = hi ? 16 : 0;

  v8f acc[4] = {};
  const size_t arow = (size_t)(tokBase + lrow) * DD;

  for (int k0 = 0; k0 < DD; k0 += 32) {
    BF16Frag ar, ai;
    const int kb = k0 + akoff;
    ar.q[0] = *(const uint4*)(qr_bf + arow + kb);
    ar.q[1] = *(const uint4*)(qr_bf + arow + kb + 8);
    ai.q[0] = *(const uint4*)(qi_bf + arow + kb);
    ai.q[1] = *(const uint4*)(qi_bf + arow + kb + 8);
#pragma unroll
    for (int t = 0; t < 4; t += 2) {
      const int colA = sBase + t * 16 + lrow;
      const int colB = colA + 16;
      const size_t boffA = (size_t)colA * DD + (size_t)(k0 + bkoff);
      const size_t boffB = (size_t)colB * DD + (size_t)(k0 + bkoff);
      BF16Frag bkr0, bki0, bkr1, bki1;
      bkr0.q[0] = ((const uint4*)(kr_bf + boffA))[0];
      bkr0.q[1] = ((const uint4*)(kr_bf + boffA))[1];
      bki0.q[0] = ((const uint4*)(ki_bf + boffA))[0];
      bki0.q[1] = ((const uint4*)(ki_bf + boffA))[1];
      bkr1.q[0] = ((const uint4*)(kr_bf + boffB))[0];
      bkr1.q[1] = ((const uint4*)(kr_bf + boffB))[1];
      bki1.q[0] = ((const uint4*)(ki_bf + boffB))[0];
      bki1.q[1] = ((const uint4*)(ki_bf + boffB))[1];
      __builtin_prefetch(kr_bf + boffA + 32, 0, 3);
      __builtin_prefetch(ki_bf + boffA + 32, 0, 3);
      __builtin_prefetch(kr_bf + boffB + 32, 0, 3);
      __builtin_prefetch(ki_bf + boffB + 32, 0, 3);
      acc[t]   = __builtin_amdgcn_wmma_f32_16x16x32_bf16(false, ar.v, false, bkr0.v, (short)0, acc[t],   false, false);
      acc[t+1] = __builtin_amdgcn_wmma_f32_16x16x32_bf16(false, ar.v, false, bkr1.v, (short)0, acc[t+1], false, false);
      acc[t]   = __builtin_amdgcn_wmma_f32_16x16x32_bf16(false, ai.v, false, bki0.v, (short)0, acc[t],   false, false);
      acc[t+1] = __builtin_amdgcn_wmma_f32_16x16x32_bf16(false, ai.v, false, bki1.v, (short)0, acc[t+1], false, false);
    }
  }
  const int moff = hi ? 8 : 0;
  float qm[8];
#pragma unroll
  for (int r = 0; r < 8; ++r) qm[r] = qmag[tokBase + moff + r];
#pragma unroll
  for (int t = 0; t < 4; ++t) {
    const int col = sBase + t * 16 + lrow;
    const float km = kmag[col];
#pragma unroll
    for (int r = 0; r < 8; ++r) {
      const size_t o = (size_t)(tokBase + moff + r) * SS + col;
      scores[o] = acc[t][r] / (qm[r] * km + EPSF);
    }
  }
}

// ---------------------------------------------------------------------------
// 7) Fused: top-8 -> softmax -> value gather -> complex RMS norm -> out
__global__ __launch_bounds__(256) void topk_out_kernel(const float* __restrict__ scores,
    const float* __restrict__ values, const float* __restrict__ gamma, float* __restrict__ out) {
  const int tok = blockIdx.x, tid = threadIdx.x;
  __shared__ float s_sc[SS];
  __shared__ float s_rv[256];
  __shared__ int   s_ri[256];
  __shared__ float s_topv[KTOP];
  __shared__ int   s_topi[KTOP];

  const float* srow = scores + (size_t)tok * SS;
#pragma unroll
  for (int j = 0; j < SS / 256; ++j) s_sc[tid + 256 * j] = srow[tid + 256 * j];
  __syncthreads();

  for (int p = 0; p < KTOP; ++p) {
    float best = -3.4e38f; int bi = 0;
#pragma unroll
    for (int j = 0; j < SS / 256; ++j) {
      int idx = tid + 256 * j;
      float v = s_sc[idx];
      if (v > best) { best = v; bi = idx; }     // ascending idx -> lowest-index ties
    }
    s_rv[tid] = best; s_ri[tid] = bi;
    __syncthreads();
    for (int off = 128; off > 0; off >>= 1) {
      if (tid < off) {
        float v1 = s_rv[tid], v2 = s_rv[tid + off];
        int   i1 = s_ri[tid], i2 = s_ri[tid + off];
        if (v2 > v1 || (v2 == v1 && i2 < i1)) { s_rv[tid] = v2; s_ri[tid] = i2; }
      }
      __syncthreads();
    }
    if (tid == 0) { s_topv[p] = s_rv[0]; s_topi[p] = s_ri[0]; s_sc[s_ri[0]] = -3.4e38f; }
    __syncthreads();
  }

  // softmax over top-8 (descending -> max is s_topv[0])
  float attn[KTOP]; float ssum = 0.f;
  const float mx = s_topv[0];
#pragma unroll
  for (int k = 0; k < KTOP; ++k) { attn[k] = expf(s_topv[k] - mx); ssum += attn[k]; }
  const float isum = 1.f / ssum;
#pragma unroll
  for (int k = 0; k < KTOP; ++k) attn[k] *= isum;

  // weighted value gather: this thread owns d0..d0+3
  const int d0 = tid * (DD / 256);
  float orr[4] = {0,0,0,0}, oii[4] = {0,0,0,0};
  for (int k = 0; k < KTOP; ++k) {
    const float* vrow = values + (size_t)s_topi[k] * (DD * 2);
    const float wgt = attn[k];
#pragma unroll
    for (int j = 0; j < 4; ++j) {
      float2 v = *(const float2*)(vrow + 2 * (d0 + j));
      orr[j] += wgt * v.x; oii[j] += wgt * v.y;
    }
  }
  // complex RMS norm
  float msum = 0.f;
#pragma unroll
  for (int j = 0; j < 4; ++j) msum += orr[j] * orr[j] + oii[j] * oii[j];
  s_rv[tid] = msum; __syncthreads();
  for (int off = 128; off > 0; off >>= 1) {
    if (tid < off) s_rv[tid] += s_rv[tid + off];
    __syncthreads();
  }
  const float rinv = 1.f / sqrtf(s_rv[0] / (float)DD + 1e-8f);
  float* orow = out + (size_t)tok * (DD * 2);
#pragma unroll
  for (int j = 0; j < 4; ++j) {
    const int d = d0 + j;
    const float g = gamma[d] * rinv;
    orow[2 * d]     = orr[j] * g;
    orow[2 * d + 1] = oii[j] * g;
  }
}

// ---------------------------------------------------------------------------
extern "C" void kernel_launch(void* const* d_in, const int* in_sizes, int n_in,
                              void* d_out, int out_size, void* d_ws, size_t ws_size,
                              hipStream_t stream) {
  (void)in_sizes; (void)n_in; (void)out_size; (void)ws_size;
  const float* x      = (const float*)d_in[0];   // [4,2048,1024,2]
  const float* Wr     = (const float*)d_in[1];   // [1024,1024]
  const float* Wi     = (const float*)d_in[2];   // [1024,1024]
  const float* keys   = (const float*)d_in[3];   // [4096,1024,2]
  const float* values = (const float*)d_in[4];   // [4096,1024,2]
  const float* gamma  = (const float*)d_in[5];   // [1024]
  float* out = (float*)d_out;                    // [4,2048,1024,2]

  char* ws = (char*)d_ws;
  constexpr size_t MB = 1ull << 20;
  unsigned short* qr_bf  = (unsigned short*)(ws + 0 * MB);   // N*D bf16 = 16MB
  unsigned short* qi_bf  = (unsigned short*)(ws + 16 * MB);  // 16MB
  unsigned short* wr_bf  = (unsigned short*)(ws + 32 * MB);  // D*D bf16 = 2MB
  unsigned short* wi_bf  = (unsigned short*)(ws + 34 * MB);  // 2MB
  unsigned short* win_bf = (unsigned short*)(ws + 36 * MB);  // 2MB (-Wi)
  unsigned short* kr_bf  = (unsigned short*)(ws + 38 * MB);  // S*D bf16 = 8MB
  unsigned short* ki_bf  = (unsigned short*)(ws + 46 * MB);  // 8MB
  unsigned short* xr_bf  = (unsigned short*)(ws + 54 * MB);  // N*D bf16 = 16MB
  unsigned short* xi_bf  = (unsigned short*)(ws + 70 * MB);  // 16MB
  float* kmag   = (float*)(ws + 86 * MB);                    // 16KB
  float* qmag   = (float*)(ws + 86 * MB + (64ull << 10));    // 32KB
  float* scores = (float*)(ws + 87 * MB);                    // N*S f32 = 128MB

  conv_w_kernel<<<(DD * DD + 255) / 256, 256, 0, stream>>>(Wr, Wi, wr_bf, wi_bf, win_bf, DD * DD);
  conv_keys_kernel<<<SS, 256, 0, stream>>>(keys, kr_bf, ki_bf, kmag);
  conv_x_kernel<<<(NN * DD) / 256, 256, 0, stream>>>(x, xr_bf, xi_bf);
  gemm_q_kernel<<<(NN / 16) * 16 / 8, 256, 0, stream>>>(xr_bf, xi_bf, wr_bf, wi_bf, win_bf,
                                                        qr_bf, qi_bf);                     // 1024 blocks
  qmag_kernel<<<NN, 256, 0, stream>>>(qr_bf, qi_bf, qmag);
  gemm_scores_kernel<<<(NN / 16) * (SS / 64) / 8, 256, 0, stream>>>(qr_bf, qi_bf, kr_bf, ki_bf,
                                                                    qmag, kmag, scores);   // 4096 blocks
  topk_out_kernel<<<NN, 256, 0, stream>>>(scores, values, gamma, out);
}
